// DistanceFusionBlock_6073083757180
// MI455X (gfx1250) — compile-verified, matrix operand order checked by reading, not run on hardware
//
#include <hip/hip_runtime.h>
#include <hip/hip_bf16.h>
#include <math.h>

// ---------------------------------------------------------------------------
// Problem dims (compile-time, from the reference)
// ---------------------------------------------------------------------------
#define BB 2
#define NN 256
#define DD 512
#define HH 2048
#define RR (BB * NN)   // 512 rows of tokens per branch

typedef __attribute__((ext_vector_type(16))) __bf16 v16bf;
typedef __attribute__((ext_vector_type(8)))  __bf16 v8bf;
typedef __attribute__((ext_vector_type(8)))  float  v8f;

union V16U { v16bf v; v8bf h[2]; };

// ---------------------------------------------------------------------------
// Kernel: zero a float buffer
// ---------------------------------------------------------------------------
__global__ void zero_f32(float* p, int n) {
    int i = blockIdx.x * blockDim.x + threadIdx.x;
    if (i < n) p[i] = 0.0f;
}

// ---------------------------------------------------------------------------
// Kernel: pairwise L1 distance row sums.
// One block per (b, i): thread j computes sum_d |xv[b,i,d] - xa[b,j,d]|.
// dv_sum[b*N+i] = sum_j dist ; da_sum[b*N+j] += dist (atomic across blocks).
// ---------------------------------------------------------------------------
__global__ __launch_bounds__(256)
void dist_kernel(const float* __restrict__ xv, const float* __restrict__ xa,
                 float* __restrict__ dv_sum, float* __restrict__ da_sum) {
    const int b = blockIdx.x / NN;
    const int i = blockIdx.x % NN;
    const int tid = threadIdx.x;           // 256 threads; tid == j

    __shared__ float sv[DD];
    const float* vrow = xv + (size_t)(b * NN + i) * DD;
    for (int d = tid; d < DD; d += 256) sv[d] = vrow[d];
    __syncthreads();

    const float* arow = xa + (size_t)(b * NN + tid) * DD;
    float s = 0.0f;
    #pragma unroll 4
    for (int d = 0; d < DD; d += 4) {
        float4 a4 = *(const float4*)(arow + d);
        s += fabsf(sv[d + 0] - a4.x);
        s += fabsf(sv[d + 1] - a4.y);
        s += fabsf(sv[d + 2] - a4.z);
        s += fabsf(sv[d + 3] - a4.w);
    }
    atomicAdd(&da_sum[b * NN + tid], s);

    __shared__ float red[256];
    red[tid] = s;
    __syncthreads();
    for (int off = 128; off > 0; off >>= 1) {
        if (tid < off) red[tid] += red[tid + off];
        __syncthreads();
    }
    if (tid == 0) dv_sum[b * NN + i] = red[0];
}

// ---------------------------------------------------------------------------
// Kernel: A = bf16(x * row_sum / 256)  (row-scaled token matrix, R x D)
// ---------------------------------------------------------------------------
__global__ void scale_to_bf16(const float* __restrict__ x,
                              const float* __restrict__ row_sum,
                              __bf16* __restrict__ out, int n) {
    int i = blockIdx.x * blockDim.x + threadIdx.x;
    if (i < n) {
        int r = i / DD;
        out[i] = (__bf16)(x[i] * row_sum[r] * (1.0f / 256.0f));
    }
}

// ---------------------------------------------------------------------------
// Kernel: transposed weight convert  Wt[n*K + k] = bf16(W[k*N + n])
// W is (K x N) row-major; Wt is (N x K) row-major so the GEMM B-fragment
// loads become contiguous b128 loads.
// ---------------------------------------------------------------------------
__global__ void wconv_t(const float* __restrict__ W, __bf16* __restrict__ Wt,
                        int K, int N) {
    int i = blockIdx.x * blockDim.x + threadIdx.x;
    if (i < K * N) {
        int n = i / K;
        int k = i % K;
        Wt[i] = (__bf16)(W[(size_t)k * N + n]);
    }
}

// ---------------------------------------------------------------------------
// WMMA GEMM:  out[M x N] = act(A[M x K] @ Bt[N x K]^T + bias)
// A, Bt in bf16 row-major (Bt stores B transposed).
//
// Tiling: each wave computes a 16 x (16*NT) output strip (NT accumulators),
// reusing its A fragment across NT WMMAs per K-step. 8 waves per block cover
// 16 rows x 128*NT cols. K stepped by 32 via v_wmma_f32_16x16x32_bf16 with
// f32 accumulation.
//
// Per-lane fragment layout (ISA 16-bit A 16x32): lane = 16*half + m;
// packed elems 0..7  hold K = k0 + 8*half + 0..7
// packed elems 8..15 hold K = k0 + 16 + 8*half + 0..7
// -> two contiguous 16-byte loads per matrix per step. B mirrors A with
// n = lane%16 over the transposed weight rows.
// ---------------------------------------------------------------------------
template<int NT, bool GELU, bool OUTBF16>
__global__ __launch_bounds__(256)
void wmma_gemm(const __bf16* __restrict__ A, const __bf16* __restrict__ Bt,
               const float* __restrict__ bias, void* __restrict__ outp,
               int M, int N, int K, int ldOut) {
    const int lane = threadIdx.x & 31;
    const int wave = threadIdx.x >> 5;
    const int half = lane >> 4;
    const int l16  = lane & 15;

    const int m0 = blockIdx.x * 16;                       // 16 rows per block
    const int n0 = (blockIdx.y * 8 + wave) * 16 * NT;     // 16*NT cols per wave
    if (m0 >= M || n0 >= N) return;

    const __bf16* arow = A + (size_t)(m0 + l16) * K;      // lane's A row (M=l16)
    const __bf16* brow[NT];
    #pragma unroll
    for (int t = 0; t < NT; ++t)                          // lane's B cols (N=l16)
        brow[t] = Bt + (size_t)(n0 + t * 16 + l16) * K;
    const int koff = 8 * half;

    v8f c[NT];
    #pragma unroll
    for (int t = 0; t < NT; ++t) c[t] = (v8f){};

    for (int k0 = 0; k0 < K; k0 += 32) {
        V16U a;
        a.h[0] = *(const v8bf*)(arow + k0 + koff);
        a.h[1] = *(const v8bf*)(arow + k0 + 16 + koff);
        #pragma unroll
        for (int t = 0; t < NT; ++t) {
            V16U b;
            b.h[0] = *(const v8bf*)(brow[t] + k0 + koff);
            b.h[1] = *(const v8bf*)(brow[t] + k0 + 16 + koff);
            c[t] = __builtin_amdgcn_wmma_f32_16x16x32_bf16(
                       /*neg_a=*/false, a.v, /*neg_b=*/false, b.v,
                       /*c_mod=*/(short)0, c[t],
                       /*reuse_a=*/false, /*reuse_b=*/false);
        }
    }

    // Epilogue: C/D f32 layout: VGPR r -> M = r + 8*half, N = lane%16.
    #pragma unroll
    for (int t = 0; t < NT; ++t) {
        const int n = n0 + t * 16 + l16;
        const float bn = bias[n];
        #pragma unroll
        for (int r = 0; r < 8; ++r) {
            const int m = m0 + r + 8 * half;
            float x = c[t][r] + bn;
            if (GELU) x = 0.5f * x * (1.0f + erff(x * 0.70710678118654752f));
            if (OUTBF16)
                ((__bf16*)outp)[(size_t)m * ldOut + n] = (__bf16)x;
            else
                ((float*)outp)[(size_t)m * ldOut + n] = x;
        }
    }
}

// ---------------------------------------------------------------------------
// Launch
// ---------------------------------------------------------------------------
extern "C" void kernel_launch(void* const* d_in, const int* in_sizes, int n_in,
                              void* d_out, int out_size, void* d_ws, size_t ws_size,
                              hipStream_t stream) {
    const float* x_v  = (const float*)d_in[0];
    const float* x_a  = (const float*)d_in[1];
    const float* W1v  = (const float*)d_in[2];
    const float* b1v  = (const float*)d_in[3];
    const float* W1a  = (const float*)d_in[4];
    const float* b1a  = (const float*)d_in[5];
    const float* Wmv  = (const float*)d_in[6];
    const float* bmv  = (const float*)d_in[7];
    const float* Wma  = (const float*)d_in[8];
    const float* bma  = (const float*)d_in[9];
    const float* Wout = (const float*)d_in[10];
    const float* bout = (const float*)d_in[11];
    float* out = (float*)d_out;

    // Workspace layout (256B-aligned chunks)
    char* ws = (char*)d_ws;
    size_t off = 0;
    auto alloc = [&](size_t bytes) {
        void* p = ws + off;
        off += (bytes + 255) & ~(size_t)255;
        return p;
    };
    float*  dv_sum = (float*)alloc(RR * sizeof(float));
    float*  da_sum = (float*)alloc(RR * sizeof(float));
    __bf16* Av     = (__bf16*)alloc((size_t)RR * DD * 2);
    __bf16* Aa     = (__bf16*)alloc((size_t)RR * DD * 2);
    __bf16* W1v_t  = (__bf16*)alloc((size_t)HH * DD * 2);   // [H x D]
    __bf16* W1a_t  = (__bf16*)alloc((size_t)HH * DD * 2);
    __bf16* Wmv_t  = (__bf16*)alloc((size_t)DD * HH * 2);   // [D x H]
    __bf16* Wma_t  = (__bf16*)alloc((size_t)DD * HH * 2);
    __bf16* Wout_t = (__bf16*)alloc((size_t)DD * 2 * DD * 2); // [D x 2D]
    __bf16* Hv     = (__bf16*)alloc((size_t)RR * HH * 2);
    __bf16* Ha     = (__bf16*)alloc((size_t)RR * HH * 2);
    __bf16* Comb   = (__bf16*)alloc((size_t)RR * 2 * DD * 2); // [R x 2D]
    (void)ws_size;

    // 1. zero distance accumulators
    zero_f32<<<(2 * RR + 255) / 256, 256, 0, stream>>>(dv_sum, 2 * RR);

    // 2. pairwise L1 distance row/col sums
    dist_kernel<<<BB * NN, 256, 0, stream>>>(x_v, x_a, dv_sum, da_sum);

    // 3. bf16 prep: scaled activations + transposed weights
    {
        int n = RR * DD;
        scale_to_bf16<<<(n + 255) / 256, 256, 0, stream>>>(x_v, dv_sum, Av, n);
        scale_to_bf16<<<(n + 255) / 256, 256, 0, stream>>>(x_a, da_sum, Aa, n);
    }
    {
        int n = DD * HH;
        wconv_t<<<(n + 255) / 256, 256, 0, stream>>>(W1v, W1v_t, DD, HH);
        wconv_t<<<(n + 255) / 256, 256, 0, stream>>>(W1a, W1a_t, DD, HH);
        wconv_t<<<(n + 255) / 256, 256, 0, stream>>>(Wmv, Wmv_t, HH, DD);
        wconv_t<<<(n + 255) / 256, 256, 0, stream>>>(Wma, Wma_t, HH, DD);
        int n2 = 2 * DD * DD;
        wconv_t<<<(n2 + 255) / 256, 256, 0, stream>>>(Wout, Wout_t, 2 * DD, DD);
    }

    // 4. GEMM1 (+bias+GELU) -> bf16 hidden activations  [R x H]
    {
        dim3 grid(RR / 16, HH / 256);   // NT=2: 256 cols per block
        wmma_gemm<2, true, true><<<grid, 256, 0, stream>>>(Av, W1v_t, b1v, Hv, RR, HH, DD, HH);
        wmma_gemm<2, true, true><<<grid, 256, 0, stream>>>(Aa, W1a_t, b1a, Ha, RR, HH, DD, HH);
    }

    // 5. GEMM2 (+bias) -> bf16 halves of concat buffer  [R x 2D]
    {
        dim3 grid(RR / 16, DD / 256);
        wmma_gemm<2, false, true><<<grid, 256, 0, stream>>>(Hv, Wmv_t, bmv, Comb,      RR, DD, HH, 2 * DD);
        wmma_gemm<2, false, true><<<grid, 256, 0, stream>>>(Ha, Wma_t, bma, Comb + DD, RR, DD, HH, 2 * DD);
    }

    // 6. GEMM3 (+bias) -> f32 final output  [R x D]
    {
        dim3 grid(RR / 16, DD / 256);
        wmma_gemm<2, false, false><<<grid, 256, 0, stream>>>(Comb, Wout_t, bout, out, RR, DD, 2 * DD, DD);
    }
}
